// ShiftLayer_2972117368844
// MI455X (gfx1250) — matrix operand order repaired and checked by reading, not compiled, not run on hardware
//
#include <hip/hip_runtime.h>
#include <hip/hip_bf16.h>
#include <math.h>

typedef _Float16 v8h  __attribute__((ext_vector_type(8)));
typedef _Float16 v16h __attribute__((ext_vector_type(16)));
typedef float    v8f  __attribute__((ext_vector_type(8)));
typedef unsigned int u32x4 __attribute__((ext_vector_type(4)));
typedef int      i32x4 __attribute__((ext_vector_type(4)));
typedef int      i32x8 __attribute__((ext_vector_type(8)));

#define CONCAT16(lo, hi) __builtin_shufflevector((lo), (hi), \
    0,1,2,3,4,5,6,7,8,9,10,11,12,13,14,15)

constexpr int   CC    = 32, HH = 80, WW = 80, KK = 4;
constexpr int   Hp    = HH - KK + 1;   // 77
constexpr int   Wp    = WW - KK + 1;   // 77
constexpr int   P     = Hp * Wp;       // 5929 patches
constexpr int   D     = CC * KK * KK;  // 512 = 16 k-steps of 32
constexpr int   NT    = 372;           // candidate 16-col tiles after padding
constexpr int   P_pad = NT * 16;       // 5952
constexpr int   NB    = P_pad / 32;    // 186 query blocks of 32 rows
constexpr float EPSV    = 1e-6f;
constexpr float NEG_INF = -1e9f;

#if defined(__has_builtin)
#if __has_builtin(__builtin_amdgcn_tensor_load_to_lds) && \
    __has_builtin(__builtin_amdgcn_s_wait_tensorcnt)
#define USE_TDM 1
#endif
#endif

static __device__ __forceinline__ v8f wmma_f16(v16h a, v16h b, v8f c) {
  // v_wmma_f32_16x16x32_f16  (neg_a, A, neg_b, B, c_mod, C, reuse_a, reuse_b)
  return __builtin_amdgcn_wmma_f32_16x16x32_f16(false, a, false, b, (short)0, c,
                                                false, false);
}

#ifdef USE_TDM
// 1-D TDM copy of `elems` 2-byte elements from contiguous global -> LDS.
// D# bitfields per CDNA5 ISA.
// This toolchain exposes the 6-arg builtin:
//   (u32x4 group0, i32x8 group1, i32x4 group2, i32x4 group3, i32x8, i32 cpol)
static __device__ __forceinline__ void tdm_copy_1d_h16(unsigned lds_byte_addr,
                                                       unsigned long long gaddr,
                                                       unsigned elems) {
  u32x4 g0;
  g0[0] = 1u;                                          // count=1, user descriptor
  g0[1] = lds_byte_addr;                               // lds_addr
  g0[2] = (unsigned)(gaddr & 0xFFFFFFFFull);           // global_addr[31:0]
  g0[3] = (unsigned)((gaddr >> 32) & 0x1FFFFFFull) |   // global_addr[56:32]
          (2u << 30);                                  // type=2 ("image")
  i32x8 g1;
  g1[0] = (int)(1u << 16);                             // data_size=1 -> 2 bytes
  g1[1] = (int)((elems & 0xFFFFu) << 16);              // tensor_dim0[15:0]
  g1[2] = (int)(elems >> 16);                          // tensor_dim0[31:16]
  g1[3] = (int)((elems & 0xFFFFu) << 16);              // tile_dim0 (tensor_dim1=0)
  g1[4] = 0;                                           // tile_dim1=0, tile_dim2=0
  g1[5] = (int)elems;                                  // tensor_dim0_stride[31:0]
  g1[6] = 0;
  g1[7] = 0;
  i32x4 z4 = {0, 0, 0, 0};
  i32x8 z8 = {0, 0, 0, 0, 0, 0, 0, 0};
  __builtin_amdgcn_tensor_load_to_lds(g0, g1, z4, z4, z8, 0);
}
#endif

// ---------------------------------------------------------------------------
// Kernel 1: extract patches -> hi/lo f16 planes, inv-norms, excl & valid masks
// grid = P_pad blocks, 64 threads
// ---------------------------------------------------------------------------
__global__ void prep_kernel(const float* __restrict__ low,
                            const float* __restrict__ high,
                            const int*   __restrict__ mask,
                            _Float16* __restrict__ chi, _Float16* __restrict__ clo,
                            _Float16* __restrict__ qhi, _Float16* __restrict__ qlo,
                            float* __restrict__ invn, int* __restrict__ excl,
                            float* __restrict__ validf) {
  __shared__ float red[64];
  const int p = blockIdx.x;
  const int t = threadIdx.x;
  const size_t row = (size_t)p * D;

  if (p >= P) {  // padding rows: zero data, never selectable
    for (int d = t; d < D; d += 64) {
      chi[row + d] = (_Float16)0.f; clo[row + d] = (_Float16)0.f;
      qhi[row + d] = (_Float16)0.f; qlo[row + d] = (_Float16)0.f;
    }
    if (t == 0) { invn[p] = 0.f; excl[p] = 1; validf[p] = 0.f; }
    return;
  }

  const int hp = p / Wp, wp = p % Wp;
  float ss = 0.f;
  for (int d = t; d < D; d += 64) {
    const int c = d >> 4, rem = d & 15, di = rem >> 2, dj = rem & 3;
    const int src = c * (HH * WW) + (hp + di) * WW + (wp + dj);
    const float lv = low[src], hv = high[src];
    const _Float16 lh = (_Float16)lv;
    const _Float16 ll = (_Float16)(lv - (float)lh);
    const _Float16 hh = (_Float16)hv;
    const _Float16 hl = (_Float16)(hv - (float)hh);
    chi[row + d] = lh; clo[row + d] = ll;
    qhi[row + d] = hh; qlo[row + d] = hl;
    ss += lv * lv;
  }
  red[t] = ss;
  __syncthreads();
  for (int s = 32; s > 0; s >>= 1) {
    if (t < s) red[t] += red[t + s];
    __syncthreads();
  }
  if (t == 0) {
    const float nrm = sqrtf(red[0]);
    invn[p] = 1.0f / (nrm + EPSV);
    const int M = WW + 1;  // mask is 81x81
    excl[p] = (mask[hp * M + wp] == 1) ? 1 : 0;
    const int ok = (mask[hp * M + wp]             != 0) &&
                   (mask[hp * M + wp + KK]        != 0) &&
                   (mask[(hp + KK) * M + wp]      != 0) &&
                   (mask[(hp + KK) * M + wp + KK] != 0);
    validf[p] = ok ? 1.f : 0.f;
  }
}

// ---------------------------------------------------------------------------
// Kernel 2: fused GEMM (WMMA, f16 hi/lo split) + per-query argmax
// grid = NB blocks (one 32-query tile each), 256 threads = 8 waves
// ---------------------------------------------------------------------------
__global__ __launch_bounds__(256)
void match_kernel(const _Float16* __restrict__ qhi, const _Float16* __restrict__ qlo,
                  const _Float16* __restrict__ chi, const _Float16* __restrict__ clo,
                  const float* __restrict__ invn, const int* __restrict__ excl,
                  int* __restrict__ best) {
  // 64 KB static LDS: A tile (both planes) during GEMM phase, then the same
  // space is reused for the cross-wave argmax reduction (separated by barriers).
  __shared__ __attribute__((aligned(32))) unsigned char smem[65536];
  _Float16* sQh = (_Float16*)smem;             // 32 rows x 512 halfs = 32 KB
  _Float16* sQl = (_Float16*)(smem + 32768);   // 32 KB
  float*    sMax = (float*)smem;               // reused after GEMM phase
  int*      sIdx = (int*)(smem + 4096);

  const int tid = threadIdx.x;
  const int m0  = blockIdx.x * 32;

  // ---- fill A tile (32 contiguous rows, hi+lo planes) into LDS ----
#ifdef USE_TDM
  if (tid < 32) {  // wave 0 issues both TDM descriptors
    const unsigned elems = 32u * D;  // 16384 2-byte elements per plane
    tdm_copy_1d_h16((unsigned)(size_t)sQh,
                    (unsigned long long)(size_t)(qhi + (size_t)m0 * D), elems);
    tdm_copy_1d_h16((unsigned)(size_t)sQl,
                    (unsigned long long)(size_t)(qlo + (size_t)m0 * D), elems);
    __builtin_amdgcn_s_wait_tensorcnt(0);
  }
#else
  {
    const v8h* srcH = (const v8h*)(qhi + (size_t)m0 * D);
    const v8h* srcL = (const v8h*)(qlo + (size_t)m0 * D);
    v8h* dH = (v8h*)sQh;
    v8h* dL = (v8h*)sQl;
    const int n8 = 32 * D / 8;  // 2048 16-byte chunks
    for (int i = tid; i < n8; i += 256) { dH[i] = srcH[i]; dL[i] = srcL[i]; }
  }
#endif
  __syncthreads();

  const int wave = tid >> 5;
  const int lane = tid & 31;
  const int l15  = lane & 15;   // N within tile (C/D layout), also A row base
  const int lh   = lane >> 4;   // lane half

  // running argmax; acc0[r]: row lh*8+r, acc1[r]: row 16+lh*8+r, col l15
  float rmax[16];
  int   ridx[16];
#pragma unroll
  for (int r = 0; r < 16; ++r) { rmax[r] = -INFINITY; ridx[r] = 0; }

  // A fragment bases (ISA 16-bit A layout: two 8-half chunks per lane)
  const _Float16* aH0 = sQh + l15 * D + lh * 8;
  const _Float16* aL0 = sQl + l15 * D + lh * 8;
  const _Float16* aH1 = sQh + (16 + l15) * D + lh * 8;
  const _Float16* aL1 = sQl + (16 + l15) * D + lh * 8;

  for (int ct = wave; ct < NT; ct += 8) {
    const int n = ct * 16 + l15;  // candidate column handled by this lane
    const _Float16* bH = chi + (size_t)n * D + lh * 8;
    const _Float16* bL = clo + (size_t)n * D + lh * 8;

    if (ct + 8 < NT) {  // stream next candidate tile toward the caches
      __builtin_prefetch(chi + (size_t)(n + 128) * D + lh * 16, 0, 1);
      __builtin_prefetch(clo + (size_t)(n + 128) * D + lh * 16, 0, 1);
    }

    v8f acc0 = {};
    v8f acc1 = {};
#pragma unroll 2
    for (int ks = 0; ks < 16; ++ks) {
      const int k0 = ks * 32;
      const v16h b_h = CONCAT16(*(const v8h*)(bH + k0), *(const v8h*)(bH + k0 + 16));
      const v16h b_l = CONCAT16(*(const v8h*)(bL + k0), *(const v8h*)(bL + k0 + 16));
      const v16h a0h = CONCAT16(*(const v8h*)(aH0 + k0), *(const v8h*)(aH0 + k0 + 16));
      const v16h a0l = CONCAT16(*(const v8h*)(aL0 + k0), *(const v8h*)(aL0 + k0 + 16));
      const v16h a1h = CONCAT16(*(const v8h*)(aH1 + k0), *(const v8h*)(aH1 + k0 + 16));
      const v16h a1l = CONCAT16(*(const v8h*)(aL1 + k0), *(const v8h*)(aL1 + k0 + 16));
      acc0 = wmma_f16(a0h, b_h, acc0);   // hi*hi
      acc0 = wmma_f16(a0h, b_l, acc0);   // hi*lo
      acc0 = wmma_f16(a0l, b_h, acc0);   // lo*hi
      acc1 = wmma_f16(a1h, b_h, acc1);
      acc1 = wmma_f16(a1h, b_l, acc1);
      acc1 = wmma_f16(a1l, b_h, acc1);
    }

    const float s  = invn[n];
    const bool  ex = (excl[n] != 0);
#pragma unroll
    for (int r = 0; r < 8; ++r) {
      const float v0 = ex ? NEG_INF : acc0[r] * s;
      const float v1 = ex ? NEG_INF : acc1[r] * s;
      // n strictly increases per lane -> strict '>' keeps first max (argmax rule)
      if (v0 > rmax[r])     { rmax[r]     = v0; ridx[r]     = n; }
      if (v1 > rmax[8 + r]) { rmax[8 + r] = v1; ridx[8 + r] = n; }
    }
  }

  // reduce across the 16 lanes of each half (same rows, disjoint N residues)
#pragma unroll
  for (int r = 0; r < 16; ++r) {
    for (int s = 1; s < 16; s <<= 1) {
      const float ov = __shfl_xor(rmax[r], s, 32);
      const int   oi = __shfl_xor(ridx[r], s, 32);
      if (ov > rmax[r] || (ov == rmax[r] && oi < ridx[r])) { rmax[r] = ov; ridx[r] = oi; }
    }
  }

  __syncthreads();  // all waves done reading sQ -> safe to reuse LDS
  if (l15 == 0) {
#pragma unroll
    for (int r = 0; r < 8; ++r) {
      const int row0 = lh * 8 + r;
      const int row1 = 16 + lh * 8 + r;
      sMax[wave * 32 + row0] = rmax[r];
      sIdx[wave * 32 + row0] = ridx[r];
      sMax[wave * 32 + row1] = rmax[8 + r];
      sIdx[wave * 32 + row1] = ridx[8 + r];
    }
  }
  __syncthreads();

  if (tid < 32) {
    float bv = -INFINITY;
    int   bi = 0;
    for (int w = 0; w < 8; ++w) {
      const float v = sMax[w * 32 + tid];
      const int   i = sIdx[w * 32 + tid];
      if (v > bv || (v == bv && i < bi)) { bv = v; bi = i; }
    }
    const int gm = m0 + tid;
    if (gm < P) best[gm] = bi;
  }
}

// ---------------------------------------------------------------------------
// Kernel 3: gather-form overlap-add reconstruction
// ---------------------------------------------------------------------------
__global__ void reconstruct_kernel(const float* __restrict__ low,
                                   const int*   __restrict__ best,
                                   const float* __restrict__ validf,
                                   float* __restrict__ out) {
  const int idx = blockIdx.x * blockDim.x + threadIdx.x;
  if (idx >= CC * HH * WW) return;
  const int c = idx / (HH * WW);
  const int r = idx % (HH * WW);
  const int i = r / WW, j = r % WW;

  float acc = 0.f, cnt = 0.f;
#pragma unroll
  for (int di = 0; di < KK; ++di) {
    const int hp = i - di;
    if (hp < 0 || hp >= Hp) continue;
#pragma unroll
    for (int dj = 0; dj < KK; ++dj) {
      const int wq = j - dj;
      if (wq < 0 || wq >= Wp) continue;
      const int q = hp * Wp + wq;
      if (validf[q] != 0.f) {
        const int b  = best[q];
        const int bi = b / Wp, bj = b % Wp;
        acc += low[c * (HH * WW) + (bi + di) * WW + (bj + dj)];
        cnt += 1.f;
      }
    }
  }
  out[idx] = (cnt != 0.f) ? acc / (cnt + EPSV) : low[idx];
}

// ---------------------------------------------------------------------------
extern "C" void kernel_launch(void* const* d_in, const int* in_sizes, int n_in,
                              void* d_out, int out_size, void* d_ws, size_t ws_size,
                              hipStream_t stream) {
  (void)in_sizes; (void)n_in; (void)out_size; (void)ws_size;

  const float* low  = (const float*)d_in[0];
  const float* high = (const float*)d_in[1];
  const int*   mask = (const int*)d_in[2];
  float* out = (float*)d_out;

  char* ws = (char*)d_ws;
  const size_t plane = (size_t)P_pad * D * sizeof(_Float16);  // ~6.1 MB each
  _Float16* chi = (_Float16*)(ws + 0 * plane);
  _Float16* clo = (_Float16*)(ws + 1 * plane);
  _Float16* qhi = (_Float16*)(ws + 2 * plane);
  _Float16* qlo = (_Float16*)(ws + 3 * plane);
  float* invn   = (float*)(ws + 4 * plane);
  int*   excl   = (int*)(invn + P_pad);
  float* validf = (float*)(excl + P_pad);
  int*   best   = (int*)(validf + P_pad);

  prep_kernel<<<P_pad, 64, 0, stream>>>(low, high, mask, chi, clo, qhi, qlo,
                                        invn, excl, validf);
  match_kernel<<<NB, 256, 0, stream>>>(qhi, qlo, chi, clo, invn, excl, best);
  const int npix = CC * HH * WW;
  reconstruct_kernel<<<(npix + 255) / 256, 256, 0, stream>>>(low, best, validf, out);
}
